// StructureAwareSSM_13151189860698
// MI455X (gfx1250) — compile-verified, matrix operand order checked by reading, not run on hardware
//
#include <hip/hip_runtime.h>
#include <math.h>

#define D_MODEL 192
#define D_INNER 384
#define DT_RANK 12
#define NB 4
#define HH 128
#define WWID 128
#define LL (HH*WWID)       // 16384
#define NTOK (NB*LL)       // 65536
#define XZ_E (2*D_INNER)   // 768
#define KPROJ (DT_RANK+2)  // 14 (dts | Bs | Cs), D_STATE=1

typedef __attribute__((ext_vector_type(8)))  _Float16 v8h;
typedef __attribute__((ext_vector_type(16))) _Float16 v16h;
typedef __attribute__((ext_vector_type(8)))  float    v8f;

// ---------------------------------------------------------------- casts
__global__ void cast_f32_to_f16(const float* __restrict__ src,
                                _Float16* __restrict__ dst, int n) {
  int i = blockIdx.x * blockDim.x + threadIdx.x;
  if (i < n) dst[i] = (_Float16)src[i];
}

// ---------------------------------------------------------------- WMMA GEMM
// C[m][n] = sum_k A[m*K+k] * Bm[n*K+k]   (both operands K-contiguous, "NT").
// Register-blocked: each wave32 produces a 32x64 output block = 2x4 tiles of
// 16x16, i.e. 8 independent v_wmma_f32_16x16x32_f16 per K=32 step. A fragments
// are reused 4x, B fragments 2x -> 12 b128 loads per 8 WMMA (1.5 loads/wmma)
// instead of 4 loads/wmma unblocked; 8 independent accumulator chains keep the
// XDL pipe busy with no D->A/B hazards inside a step.
// Fragment layout per CDNA5 ISA 7.12.2 (16-bit A 16x32): lane<16 -> row=lane,
// K chunks {0..7} u {16..23}; lane>=16 -> row=lane-16, K chunks {8..15} u
// {24..31}. B mirrored with N across lanes. C/D: VGPR j -> M = j+8*(lane>=16),
// N = lane&15.
__global__ void gemm_f16_nt_b2x4(const _Float16* __restrict__ A,
                                 const _Float16* __restrict__ Bm,
                                 float* __restrict__ C, int M, int N, int K) {
  int wid = (blockIdx.x * blockDim.x + threadIdx.x) >> 5;
  int nbn = N >> 6;                            // 64-wide N blocks
  if (wid >= (M >> 5) * nbn) return;           // uniform per wave: EXEC all-1s
  int mb = wid / nbn, nb = wid - mb * nbn;
  int lane = threadIdx.x & 31;
  int r  = lane & 15;
  int ks = (lane >> 4) << 3;                   // 0 or 8
  const _Float16* arow0 = A + (size_t)(mb * 32 + r) * K;
  const _Float16* arow1 = arow0 + (size_t)16 * K;
  const _Float16* brow0 = Bm + (size_t)(nb * 64 + r) * K;
  const _Float16* brow1 = brow0 + (size_t)16 * K;
  const _Float16* brow2 = brow0 + (size_t)32 * K;
  const _Float16* brow3 = brow0 + (size_t)48 * K;

  v8f acc00 = {}, acc01 = {}, acc02 = {}, acc03 = {};
  v8f acc10 = {}, acc11 = {}, acc12 = {}, acc13 = {};

  for (int k0 = 0; k0 < K; k0 += 32) {
    v8h a0l = *(const v8h*)(arow0 + k0 + ks);
    v8h a0h = *(const v8h*)(arow0 + k0 + 16 + ks);
    v8h a1l = *(const v8h*)(arow1 + k0 + ks);
    v8h a1h = *(const v8h*)(arow1 + k0 + 16 + ks);
    v8h b0l = *(const v8h*)(brow0 + k0 + ks);
    v8h b0h = *(const v8h*)(brow0 + k0 + 16 + ks);
    v8h b1l = *(const v8h*)(brow1 + k0 + ks);
    v8h b1h = *(const v8h*)(brow1 + k0 + 16 + ks);
    v8h b2l = *(const v8h*)(brow2 + k0 + ks);
    v8h b2h = *(const v8h*)(brow2 + k0 + 16 + ks);
    v8h b3l = *(const v8h*)(brow3 + k0 + ks);
    v8h b3h = *(const v8h*)(brow3 + k0 + 16 + ks);
    v16h a0 = __builtin_shufflevector(a0l, a0h, 0,1,2,3,4,5,6,7,8,9,10,11,12,13,14,15);
    v16h a1 = __builtin_shufflevector(a1l, a1h, 0,1,2,3,4,5,6,7,8,9,10,11,12,13,14,15);
    v16h b0 = __builtin_shufflevector(b0l, b0h, 0,1,2,3,4,5,6,7,8,9,10,11,12,13,14,15);
    v16h b1 = __builtin_shufflevector(b1l, b1h, 0,1,2,3,4,5,6,7,8,9,10,11,12,13,14,15);
    v16h b2 = __builtin_shufflevector(b2l, b2h, 0,1,2,3,4,5,6,7,8,9,10,11,12,13,14,15);
    v16h b3 = __builtin_shufflevector(b3l, b3h, 0,1,2,3,4,5,6,7,8,9,10,11,12,13,14,15);
    acc00 = __builtin_amdgcn_wmma_f32_16x16x32_f16(false, a0, false, b0, (short)0, acc00, false, false);
    acc01 = __builtin_amdgcn_wmma_f32_16x16x32_f16(false, a0, false, b1, (short)0, acc01, false, false);
    acc02 = __builtin_amdgcn_wmma_f32_16x16x32_f16(false, a0, false, b2, (short)0, acc02, false, false);
    acc03 = __builtin_amdgcn_wmma_f32_16x16x32_f16(false, a0, false, b3, (short)0, acc03, false, false);
    acc10 = __builtin_amdgcn_wmma_f32_16x16x32_f16(false, a1, false, b0, (short)0, acc10, false, false);
    acc11 = __builtin_amdgcn_wmma_f32_16x16x32_f16(false, a1, false, b1, (short)0, acc11, false, false);
    acc12 = __builtin_amdgcn_wmma_f32_16x16x32_f16(false, a1, false, b2, (short)0, acc12, false, false);
    acc13 = __builtin_amdgcn_wmma_f32_16x16x32_f16(false, a1, false, b3, (short)0, acc13, false, false);
  }

  int rsel = (lane >> 4) << 3;                 // 0 or 8
  v8f accs[2][4] = {{acc00, acc01, acc02, acc03}, {acc10, acc11, acc12, acc13}};
#pragma unroll
  for (int mi = 0; mi < 2; ++mi) {
#pragma unroll
    for (int ni = 0; ni < 4; ++ni) {
      int crow = mb * 32 + mi * 16 + rsel;
      int ccol = nb * 64 + ni * 16 + r;
#pragma unroll
      for (int j = 0; j < 8; ++j)
        C[(size_t)(crow + j) * N + ccol] = accs[mi][ni][j];
    }
  }
}

// ------------------------------------------- depthwise 3x3 (zero pad) + SiLU
// reads xi part of xz (token-major), writes xs channel-major [b*384+d][L]
__global__ void dwconv3_silu(const float* __restrict__ xz,
                             const float* __restrict__ w,
                             const float* __restrict__ bias,
                             float* __restrict__ xs) {
  int bd = blockIdx.y; int b = bd / D_INNER; int d = bd - b * D_INNER;
  int l = blockIdx.x * blockDim.x + threadIdx.x;
  int h = l >> 7, wc = l & (WWID - 1);
  float acc = 0.f;
#pragma unroll
  for (int kh = 0; kh < 3; ++kh) {
    int ih = h + kh - 1;
    if (ih < 0 || ih >= HH) continue;
#pragma unroll
    for (int kw = 0; kw < 3; ++kw) {
      int iw = wc + kw - 1;
      if (iw < 0 || iw >= WWID) continue;
      acc += xz[((size_t)((b * HH + ih) * WWID + iw)) * XZ_E + d] * w[d * 9 + kh * 3 + kw];
    }
  }
  acc += bias[d];
  acc = acc / (1.f + expf(-acc));            // SiLU
  xs[(size_t)bd * LL + l] = acc;
}

// ----------------------------------------- x_dbl[b][k][l] = x_proj_w @ xs
__global__ void xproj_kernel(const float* __restrict__ xpw,
                             const float* __restrict__ xs,
                             float* __restrict__ xdbl) {
  int bk = blockIdx.y; int b = bk / KPROJ; int k = bk - b * KPROJ;
  int l = blockIdx.x * blockDim.x + threadIdx.x;
  const float* xsb  = xs + (size_t)b * D_INNER * LL + l;
  const float* wrow = xpw + k * D_INNER;
  float acc = 0.f;
  for (int d = 0; d < D_INNER; ++d) acc += wrow[d] * xsb[(size_t)d * LL];
  xdbl[(size_t)bk * LL + l] = acc;
}

// ----------------------------------------- delta = softplus(dt_w@dts + dt_b)
__global__ void delta_kernel(const float* __restrict__ dtw,
                             const float* __restrict__ dtb,
                             const float* __restrict__ xdbl,
                             float* __restrict__ delta) {
  int bd = blockIdx.y; int b = bd / D_INNER; int d = bd - b * D_INNER;
  int l = blockIdx.x * blockDim.x + threadIdx.x;
  const float* dts = xdbl + (size_t)(b * KPROJ) * LL + l;
  float acc = dtb[d];
#pragma unroll
  for (int r = 0; r < DT_RANK; ++r) acc += dtw[d * DT_RANK + r] * dts[(size_t)r * LL];
  delta[(size_t)bd * LL + l] = (acc > 20.f) ? acc : log1pf(expf(acc));
}

// ------------------------------ chunked parallel scan: h = dA*h + dBu, D_STATE=1
// one block per (b,d); 256 threads x SEG=64 elems; Hillis-Steele over affine pairs
__global__ void scan_kernel(const float* __restrict__ delta,
                            const float* __restrict__ xdbl,
                            const float* __restrict__ xs,
                            const float* __restrict__ A_logs,
                            float* __restrict__ hs) {
  const int SEG = LL / 256;                   // 64
  int bd = blockIdx.x; int b = bd / D_INNER; int d = bd - b * D_INNER;
  int tid = threadIdx.x;
  float Ad = -expf(A_logs[d]);                // A = -exp(A_logs), state dim 1
  const float* del = delta + (size_t)bd * LL;
  const float* xsr = xs + (size_t)bd * LL;
  const float* Bs  = xdbl + (size_t)(b * KPROJ + DT_RANK) * LL;
  int base = tid * SEG;
  // pass 1: segment aggregate (a, b) of composed affine maps
  float a = 1.f, bb = 0.f;
  for (int i = 0; i < SEG; ++i) {
    float dv  = del[base + i];
    float dA  = expf(dv * Ad);
    float dBu = dv * Bs[base + i] * xsr[base + i];
    a = dA * a; bb = dA * bb + dBu;
  }
  __shared__ float sa[256], sb[256];
  sa[tid] = a; sb[tid] = bb;
  __syncthreads();
  // inclusive scan with non-commutative compose (prev then cur)
  for (int off = 1; off < 256; off <<= 1) {
    float pa = 1.f, pb = 0.f;
    float ca = sa[tid], cb = sb[tid];
    if (tid >= off) { pa = sa[tid - off]; pb = sb[tid - off]; }
    __syncthreads();
    sa[tid] = ca * pa; sb[tid] = ca * pb + cb;
    __syncthreads();
  }
  // h entering my segment (h_init = 0 -> state is just the B term)
  float h = (tid == 0) ? 0.f : sb[tid - 1];
  // pass 2: replay
  for (int i = 0; i < SEG; ++i) {
    float dv  = del[base + i];
    float dA  = expf(dv * Ad);
    float dBu = dv * Bs[base + i] * xsr[base + i];
    h = dA * h + dBu;
    hs[(size_t)bd * LL + base + i] = h;
  }
}

// ---------------- fused dilated dwconvs (replicate pad, dil 1/3/5) + y = hf*Cs + xs*Ds
__global__ void dilconv_y_kernel(const float* __restrict__ hs,
                                 const float* __restrict__ k3a,
                                 const float* __restrict__ k3b,
                                 const float* __restrict__ k3c,
                                 const float* __restrict__ alpha,
                                 const float* __restrict__ xdbl,
                                 const float* __restrict__ xs,
                                 const float* __restrict__ Ds,
                                 float* __restrict__ ybuf) {
  int bd = blockIdx.y; int b = bd / D_INNER; int d = bd - b * D_INNER;
  int l = blockIdx.x * blockDim.x + threadIdx.x;
  int h = l >> 7, wc = l & (WWID - 1);
  const float* hm = hs + (size_t)bd * LL;
  float s0 = 0.f, s1 = 0.f, s2 = 0.f;
#pragma unroll
  for (int kh = 0; kh < 3; ++kh) {
#pragma unroll
    for (int kw = 0; kw < 3; ++kw) {
      int dh = kh - 1, dw = kw - 1, wi = d * 9 + kh * 3 + kw;
      { int ih = min(max(h + dh,     0), HH - 1), iw = min(max(wc + dw,     0), WWID - 1);
        s0 += hm[ih * WWID + iw] * k3a[wi]; }
      { int ih = min(max(h + dh * 3, 0), HH - 1), iw = min(max(wc + dw * 3, 0), WWID - 1);
        s1 += hm[ih * WWID + iw] * k3b[wi]; }
      { int ih = min(max(h + dh * 5, 0), HH - 1), iw = min(max(wc + dw * 5, 0), WWID - 1);
        s2 += hm[ih * WWID + iw] * k3c[wi]; }
    }
  }
  float hf = alpha[0] * s0 + alpha[1] * s1 + alpha[2] * s2;
  float Cs = xdbl[(size_t)(b * KPROJ + DT_RANK + 1) * LL + l];
  float y  = hf * Cs + xs[(size_t)bd * LL + l] * Ds[d];
  ybuf[((size_t)b * LL + l) * D_INNER + d] = y;   // token-major for LN + GEMM
}

// -------------------------- LayerNorm over 384 ch + * silu(z) + cast to f16
__global__ void ln_silu_kernel(const float* __restrict__ ybuf,
                               const float* __restrict__ xz,
                               const float* __restrict__ g,
                               const float* __restrict__ bta,
                               _Float16* __restrict__ yh) {
  int n = blockIdx.x;
  int d = threadIdx.x;                        // blockDim = 384
  float v = ybuf[(size_t)n * D_INNER + d];
  __shared__ float s1[512], s2[512];
  s1[d] = v; s2[d] = v * v;
  if (d < 512 - D_INNER) { s1[D_INNER + d] = 0.f; s2[D_INNER + d] = 0.f; }
  __syncthreads();
  for (int s = 256; s > 0; s >>= 1) {
    if (d < s) { s1[d] += s1[d + s]; s2[d] += s2[d + s]; }
    __syncthreads();
  }
  float mu  = s1[0] * (1.f / D_INNER);
  float var = s2[0] * (1.f / D_INNER) - mu * mu;
  float val = (v - mu) * rsqrtf(var + 1e-5f) * g[d] + bta[d];
  float zv  = xz[(size_t)n * XZ_E + D_INNER + d];
  val *= zv / (1.f + expf(-zv));              // * silu(z)
  yh[(size_t)n * D_INNER + d] = (_Float16)val;
}

// ---------------------------------------------------------------- launcher
extern "C" void kernel_launch(void* const* d_in, const int* in_sizes, int n_in,
                              void* d_out, int out_size, void* d_ws, size_t ws_size,
                              hipStream_t stream) {
  const float* x      = (const float*)d_in[0];
  const float* Win    = (const float*)d_in[1];
  const float* conv_w = (const float*)d_in[2];
  const float* conv_b = (const float*)d_in[3];
  const float* xpw    = (const float*)d_in[4];
  const float* dt_w   = (const float*)d_in[5];
  const float* dt_b   = (const float*)d_in[6];
  const float* A_logs = (const float*)d_in[7];
  const float* Ds     = (const float*)d_in[8];
  const float* k3     = (const float*)d_in[9];
  const float* k3_1   = (const float*)d_in[10];
  const float* k3_2   = (const float*)d_in[11];
  const float* alpha  = (const float*)d_in[12];
  const float* ln_g   = (const float*)d_in[13];
  const float* ln_b   = (const float*)d_in[14];
  const float* Wout   = (const float*)d_in[15];
  float* out = (float*)d_out;

  char* ws = (char*)d_ws;
  size_t off = 0;
  auto carve = [&](size_t bytes) -> char* {
    char* p = ws + off;
    off = (off + bytes + 255) & ~(size_t)255;
    return p;
  };
  _Float16* xh    = (_Float16*)carve((size_t)NTOK * D_MODEL * 2);
  _Float16* winh  = (_Float16*)carve((size_t)XZ_E * D_MODEL * 2);
  _Float16* wouth = (_Float16*)carve((size_t)D_MODEL * D_INNER * 2);
  float* xz   = (float*)carve((size_t)NTOK * XZ_E * 4);
  float* xs   = (float*)carve((size_t)NTOK * D_INNER * 4);
  float* xdbl = (float*)carve((size_t)NB * KPROJ * LL * 4);
  float* dl   = (float*)carve((size_t)NTOK * D_INNER * 4);
  float* hsb  = (float*)carve((size_t)NTOK * D_INNER * 4);
  float* ybuf = (float*)carve((size_t)NTOK * D_INNER * 4);
  _Float16* yh = (_Float16*)carve((size_t)NTOK * D_INNER * 2);

  // casts to f16 for the WMMA GEMMs
  { int n = NTOK * D_MODEL;
    cast_f32_to_f16<<<(n + 255) / 256, 256, 0, stream>>>(x, xh, n); }
  { int n = XZ_E * D_MODEL;
    cast_f32_to_f16<<<(n + 255) / 256, 256, 0, stream>>>(Win, winh, n); }
  { int n = D_MODEL * D_INNER;
    cast_f32_to_f16<<<(n + 255) / 256, 256, 0, stream>>>(Wout, wouth, n); }

  // GEMM-in: xz[N,768] = x[N,192] * Win^T   (32x64 block/wave: 2048*12 waves)
  { int waves = (NTOK / 32) * (XZ_E / 64);
    gemm_f16_nt_b2x4<<<waves / 8, 256, 0, stream>>>(xh, winh, xz, NTOK, XZ_E, D_MODEL); }

  dim3 gBD(LL / 256, NB * D_INNER);
  dwconv3_silu<<<gBD, 256, 0, stream>>>(xz, conv_w, conv_b, xs);

  dim3 gBK(LL / 256, NB * KPROJ);
  xproj_kernel<<<gBK, 256, 0, stream>>>(xpw, xs, xdbl);

  delta_kernel<<<gBD, 256, 0, stream>>>(dt_w, dt_b, xdbl, dl);

  scan_kernel<<<NB * D_INNER, 256, 0, stream>>>(dl, xdbl, xs, A_logs, hsb);

  dilconv_y_kernel<<<gBD, 256, 0, stream>>>(hsb, k3, k3_1, k3_2, alpha, xdbl, xs, Ds, ybuf);

  ln_silu_kernel<<<NTOK, D_INNER, 0, stream>>>(ybuf, xz, ln_g, ln_b, yh);

  // GEMM-out: out[N,192] = y[N,384] * Wout^T  (32x64 block/wave: 2048*3 waves)
  { int waves = (NTOK / 32) * (D_MODEL / 64);
    gemm_f16_nt_b2x4<<<waves / 8, 256, 0, stream>>>(yh, wouth, out, NTOK, D_MODEL, D_INNER); }
}
